// RecurrentSetRepresentationModel_55894704390525
// MI455X (gfx1250) — compile-verified, matrix operand order checked by reading, not run on hardware
//
#include <hip/hip_runtime.h>
#include <hip/hip_bf16.h>
#include <cstdint>
#include <cstddef>

typedef __attribute__((ext_vector_type(16))) _Float16 v16h;
typedef __attribute__((ext_vector_type(8)))  float    v8f;

#define B_SZ    2048
#define T_SZ    200
#define DI_SZ   64
#define H_SZ    128
#define DU_SZ   128
#define G3      384      // 3*H (r,z,n gates)
#define DIN_MLP 25728    // T*H + DU

// ---------------------------------------------------------------------------
// WMMA fragment helpers (layouts per CDNA5 ISA 7.12.2, 16-bit 16x32 A / 32x16 B)
// half j of lane l maps to k = 8*(l>>4) + (j<8 ? j : j+8); pairs are contiguous.
// ---------------------------------------------------------------------------
__device__ __forceinline__ int kpat(int lane, int p) {
    return ((lane >> 4) << 3) + ((p < 4) ? (2 * p) : (2 * p + 8));
}

// Load a 16x32 (or 32x16 transposed-weight) f16 fragment; `row` fixed per lane.
__device__ __forceinline__ v16h load_frag_g(const _Float16* __restrict__ base,
                                            int row, int ldr, int k0, int lane) {
    v16h f;
    const _Float16* p = base + (size_t)row * (size_t)ldr + k0;
    #pragma unroll
    for (int q = 0; q < 8; ++q) {
        int k = kpat(lane, q);
        f[2 * q]     = p[k];
        f[2 * q + 1] = p[k + 1];
    }
    return f;
}

__device__ __forceinline__ v8f wmma_f16(v16h a, v16h b, v8f c) {
    return __builtin_amdgcn_wmma_f32_16x16x32_f16(false, a, false, b, (short)0, c,
                                                  false, false);
}

__device__ __forceinline__ float sigmoidf_(float x) {
    return 1.0f / (1.0f + __expf(-x));
}

// ---------------------------------------------------------------------------
// Elementwise f32 -> f16 conversion (weights; grid-strided)
// ---------------------------------------------------------------------------
__global__ __launch_bounds__(256) void cvt_f32_f16_kernel(const float* __restrict__ in,
                                                          _Float16* __restrict__ out,
                                                          size_t n) {
    size_t i = (size_t)blockIdx.x * blockDim.x + threadIdx.x;
    size_t stride = (size_t)gridDim.x * blockDim.x;
    for (; i < n; i += stride) out[i] = (_Float16)in[i];
}

// Scatter user embeddings into column 0..127 of the concatenated MLP input X0
__global__ __launch_bounds__(256) void user_to_x0_kernel(const float* __restrict__ u,
                                                         _Float16* __restrict__ X0) {
    int i = blockIdx.x * blockDim.x + threadIdx.x;
    if (i < B_SZ * DU_SZ) {
        int r = i >> 7, c = i & 127;
        X0[(size_t)r * DIN_MLP + c] = (_Float16)u[i];
    }
}

// ---------------------------------------------------------------------------
// Generic WMMA GEMM:  out[M,N] = A[M,K] * W[N,K]^T + bias
// Block = 256 threads (8 waves), tile 64x64; A staged via LDS (with f32->f16
// conversion if needed); W fragments streamed from global (L2-resident f16).
// ---------------------------------------------------------------------------
template <bool A_F16, bool OUT_F16>
__global__ __launch_bounds__(256) void wmma_gemm_kernel(
    const float* __restrict__ Af, const _Float16* __restrict__ Ah,
    const _Float16* __restrict__ W, const float* __restrict__ bias,
    float* __restrict__ Cf, _Float16* __restrict__ Ch,
    int M, int N, int K) {
    __shared__ _Float16 As[64 * 32];

    const int tid  = threadIdx.x;
    const int wid  = tid >> 5;
    const int lane = tid & 31;
    const int row0 = blockIdx.x * 64;
    const int col0 = blockIdx.y * 64;

    const int mt  = wid & 3;   // m-tile (0..3) per wave
    const int ntp = wid >> 2;  // n-tile pair (0..1) per wave

    v8f c0 = {}, c1 = {};

    const int lr = tid >> 2;         // staging row 0..63
    const int lc = (tid & 3) * 8;    // staging col {0,8,16,24}

    for (int k0 = 0; k0 < K; k0 += 32) {
        if (A_F16) {
            const _Float16* src = Ah + (size_t)(row0 + lr) * (size_t)K + k0 + lc;
            #pragma unroll
            for (int i = 0; i < 8; ++i) As[lr * 32 + lc + i] = src[i];
        } else {
            const float* src = Af + (size_t)(row0 + lr) * (size_t)K + k0 + lc;
            #pragma unroll
            for (int i = 0; i < 8; ++i) As[lr * 32 + lc + i] = (_Float16)src[i];
        }
        __syncthreads();

        v16h a;
        {
            int r = mt * 16 + (lane & 15);
            const _Float16* p = &As[r * 32];
            #pragma unroll
            for (int q = 0; q < 8; ++q) {
                int k = kpat(lane, q);
                a[2 * q]     = p[k];
                a[2 * q + 1] = p[k + 1];
            }
        }
        {
            int col = col0 + (ntp * 2 + 0) * 16 + (lane & 15);
            v16h b = load_frag_g(W, col, K, k0, lane);
            c0 = wmma_f16(a, b, c0);
        }
        {
            int col = col0 + (ntp * 2 + 1) * 16 + (lane & 15);
            v16h b = load_frag_g(W, col, K, k0, lane);
            c1 = wmma_f16(a, b, c1);
        }
        __syncthreads();
    }

    #pragma unroll
    for (int q = 0; q < 2; ++q) {
        v8f c = q ? c1 : c0;
        int col = col0 + (ntp * 2 + q) * 16 + (lane & 15);
        float bb = bias ? bias[col] : 0.0f;
        #pragma unroll
        for (int v = 0; v < 8; ++v) {
            int r = row0 + mt * 16 + ((lane >> 4) << 3) + v;
            float val = c[v] + bb;
            if (OUT_F16) Ch[(size_t)r * (size_t)N + col] = (_Float16)val;
            else         Cf[(size_t)r * (size_t)N + col] = val;
        }
    }
}

// ---------------------------------------------------------------------------
// Persistent fused GRU layer. Each block owns 32 batch rows and runs the
// whole T=200 recurrence internally (h[b] depends only on h[b]: no cross-
// block sync). w_hh B-fragments (3 n-tiles x 4 k-chunks) live in registers
// for the whole kernel; h lives in LDS as f16; hg accum spilled to LDS f32
// for the gate phase. y written as f16 directly into its slot in the
// (possibly concatenated) output matrix.
// ---------------------------------------------------------------------------
__global__ __launch_bounds__(256) void gru_layer_kernel(
    const _Float16* __restrict__ xg,    // [B*T, 384] precomputed input gates
    const _Float16* __restrict__ w_hh,  // [384, 128] f16
    const float* __restrict__ b_hh,     // [384]
    const float* __restrict__ h0,       // [B, 128] f32 (layer slice)
    _Float16* __restrict__ y_out,       // base (+col offset); elem (b,t,j) at b*stride + t*128 + j
    int y_stride) {
    __shared__ _Float16 hs[32 * H_SZ];     // 8 KB
    __shared__ float    accs[32 * G3];     // 48 KB  (total 56 KB LDS)

    const int tid  = threadIdx.x;
    const int wid  = tid >> 5;
    const int lane = tid & 31;
    const int row0 = blockIdx.x * 32;

    for (int i = tid; i < 32 * H_SZ; i += 256)
        hs[i] = (_Float16)h0[(size_t)row0 * H_SZ + i];

    // Register-resident w_hh fragments: wave wid owns n-tiles wid*3 .. wid*3+2
    v16h bf[3][4];
    #pragma unroll
    for (int i = 0; i < 3; ++i) {
        int col = (wid * 3 + i) * 16 + (lane & 15);
        #pragma unroll
        for (int kc = 0; kc < 4; ++kc)
            bf[i][kc] = load_frag_g(w_hh, col, H_SZ, kc * 32, lane);
    }

    const int gr = tid >> 3;          // gate row (0..31)
    const int gj = (tid & 7) * 16;    // gate col base (0..112)

    for (int t = 0; t < T_SZ; ++t) {
        __syncthreads();  // h ready / previous acc consumed

        // hg = h @ w_hh^T  (32 x 384, K=128)
        #pragma unroll
        for (int mt = 0; mt < 2; ++mt) {
            v16h af[4];
            #pragma unroll
            for (int kc = 0; kc < 4; ++kc) {
                int r = mt * 16 + (lane & 15);
                const _Float16* p = &hs[r * H_SZ + kc * 32];
                #pragma unroll
                for (int q = 0; q < 8; ++q) {
                    int k = kpat(lane, q);
                    af[kc][2 * q]     = p[k];
                    af[kc][2 * q + 1] = p[k + 1];
                }
            }
            #pragma unroll
            for (int i = 0; i < 3; ++i) {
                v8f c = {};
                #pragma unroll
                for (int kc = 0; kc < 4; ++kc) c = wmma_f16(af[kc], bf[i][kc], c);
                int col = (wid * 3 + i) * 16 + (lane & 15);
                #pragma unroll
                for (int v = 0; v < 8; ++v) {
                    int r = mt * 16 + ((lane >> 4) << 3) + v;
                    accs[r * G3 + col] = c[v];
                }
            }
        }
        __syncthreads();

        // Gate math: r,z,n -> h_new ; update LDS h and write y (f16)
        size_t xoff = ((size_t)(row0 + gr) * T_SZ + (size_t)t) * G3;
        #pragma unroll 4
        for (int jj = 0; jj < 16; ++jj) {
            int j = gj + jj;
            float xr = (float)xg[xoff + j];
            float xz = (float)xg[xoff + H_SZ + j];
            float xn = (float)xg[xoff + 2 * H_SZ + j];
            float hr = accs[gr * G3 + j]            + b_hh[j];
            float hz = accs[gr * G3 + H_SZ + j]     + b_hh[H_SZ + j];
            float hn = accs[gr * G3 + 2 * H_SZ + j] + b_hh[2 * H_SZ + j];
            float rg = sigmoidf_(xr + hr);
            float zg = sigmoidf_(xz + hz);
            float ng = tanhf(xn + rg * hn);
            float ho = (float)hs[gr * H_SZ + j];
            float hnew = (1.0f - zg) * ng + zg * ho;
            hs[gr * H_SZ + j] = (_Float16)hnew;
            y_out[(size_t)(row0 + gr) * (size_t)y_stride + (size_t)t * H_SZ + j] =
                (_Float16)hnew;
        }
    }
}

// ---------------------------------------------------------------------------
// Row LayerNorm (affine) + exact GELU, f32 in -> f16 out. One block per row.
// ---------------------------------------------------------------------------
__global__ __launch_bounds__(256) void ln_gelu_kernel(const float* __restrict__ Z,
                                                      const float* __restrict__ g,
                                                      const float* __restrict__ be,
                                                      _Float16* __restrict__ X,
                                                      int cols) {
    __shared__ float red[256];
    const int row = blockIdx.x, tid = threadIdx.x;
    const float* z = Z + (size_t)row * (size_t)cols;

    float s = 0.0f;
    for (int c = tid; c < cols; c += 256) s += z[c];
    red[tid] = s;
    __syncthreads();
    for (int o = 128; o > 0; o >>= 1) {
        if (tid < o) red[tid] += red[tid + o];
        __syncthreads();
    }
    float mu = red[0] / (float)cols;
    __syncthreads();

    float v = 0.0f;
    for (int c = tid; c < cols; c += 256) {
        float d = z[c] - mu;
        v += d * d;
    }
    red[tid] = v;
    __syncthreads();
    for (int o = 128; o > 0; o >>= 1) {
        if (tid < o) red[tid] += red[tid + o];
        __syncthreads();
    }
    float rstd = rsqrtf(red[0] / (float)cols + 1e-5f);

    for (int c = tid; c < cols; c += 256) {
        float xv = (z[c] - mu) * rstd * g[c] + be[c];
        float ge = 0.5f * xv * (1.0f + erff(xv * 0.70710678118654752f));
        X[(size_t)row * (size_t)cols + c] = (_Float16)ge;
    }
}

// ---------------------------------------------------------------------------
// Host orchestration
// ---------------------------------------------------------------------------
extern "C" void kernel_launch(void* const* d_in, const int* in_sizes, int n_in,
                              void* d_out, int out_size, void* d_ws, size_t ws_size,
                              hipStream_t stream) {
    (void)in_sizes; (void)n_in; (void)out_size; (void)ws_size;

    const float* user  = (const float*)d_in[0];   // [B, 128]
    const float* neigh = (const float*)d_in[1];   // [B, T, 64]
    const float* h0    = (const float*)d_in[2];   // [2, B, 128]
    const float* w_ih0 = (const float*)d_in[3];   // [384, 64]
    const float* w_hh0 = (const float*)d_in[4];   // [384, 128]
    const float* b_ih0 = (const float*)d_in[5];
    const float* b_hh0 = (const float*)d_in[6];
    const float* w_ih1 = (const float*)d_in[7];   // [384, 128]
    const float* w_hh1 = (const float*)d_in[8];   // [384, 128]
    const float* b_ih1 = (const float*)d_in[9];
    const float* b_hh1 = (const float*)d_in[10];
    const float* W0    = (const float*)d_in[11];  // [1024, 25728]
    const float* b0    = (const float*)d_in[12];
    const float* g0    = (const float*)d_in[13];
    const float* be0   = (const float*)d_in[14];
    const float* W1    = (const float*)d_in[15];  // [512, 1024]
    const float* b1    = (const float*)d_in[16];
    const float* g1    = (const float*)d_in[17];
    const float* be1   = (const float*)d_in[18];
    const float* W2    = (const float*)d_in[19];  // [128, 512]
    const float* b2    = (const float*)d_in[20];

    // ---- workspace layout (f16 staging + intermediates), 256B aligned ----
    char* ws = (char*)d_ws;
    size_t off = 0;
    auto take = [&](size_t bytes) -> void* {
        void* p = ws + off;
        off = (off + bytes + 255) & ~(size_t)255;
        return p;
    };
    _Float16* w_ih0_h = (_Float16*)take((size_t)G3 * DI_SZ * 2);
    _Float16* w_hh0_h = (_Float16*)take((size_t)G3 * H_SZ * 2);
    _Float16* w_ih1_h = (_Float16*)take((size_t)G3 * H_SZ * 2);
    _Float16* w_hh1_h = (_Float16*)take((size_t)G3 * H_SZ * 2);
    _Float16* W0_h    = (_Float16*)take((size_t)1024 * DIN_MLP * 2);
    _Float16* W1_h    = (_Float16*)take((size_t)512 * 1024 * 2);
    _Float16* W2_h    = (_Float16*)take((size_t)128 * 512 * 2);
    _Float16* xg      = (_Float16*)take((size_t)B_SZ * T_SZ * G3 * 2);   // reused both layers
    _Float16* y0      = (_Float16*)take((size_t)B_SZ * T_SZ * H_SZ * 2);
    _Float16* X0      = (_Float16*)take((size_t)B_SZ * DIN_MLP * 2);     // concat(user, y1)
    float*    Zbuf    = (float*)take((size_t)B_SZ * 1024 * 4);           // reused both LN layers
    _Float16* X1      = (_Float16*)take((size_t)B_SZ * 1024 * 2);
    _Float16* X2      = (_Float16*)take((size_t)B_SZ * 512 * 2);

    // ---- weight conversion to f16 ----
    auto cvt = [&](const float* src, _Float16* dst, size_t n) {
        int blocks = (int)((n + 2047) / 2048);
        if (blocks > 2048) blocks = 2048;
        hipLaunchKernelGGL(cvt_f32_f16_kernel, dim3(blocks), dim3(256), 0, stream,
                           src, dst, n);
    };
    cvt(w_ih0, w_ih0_h, (size_t)G3 * DI_SZ);
    cvt(w_hh0, w_hh0_h, (size_t)G3 * H_SZ);
    cvt(w_ih1, w_ih1_h, (size_t)G3 * H_SZ);
    cvt(w_hh1, w_hh1_h, (size_t)G3 * H_SZ);
    cvt(W0, W0_h, (size_t)1024 * DIN_MLP);
    cvt(W1, W1_h, (size_t)512 * 1024);
    cvt(W2, W2_h, (size_t)128 * 512);

    hipLaunchKernelGGL(user_to_x0_kernel, dim3((B_SZ * DU_SZ + 255) / 256), dim3(256),
                       0, stream, user, X0);

    const int BT = B_SZ * T_SZ;  // 409600

    // ---- layer 0: xg0 = x @ w_ih0^T + b_ih0 ; GRU recurrence -> y0 ----
    hipLaunchKernelGGL((wmma_gemm_kernel<false, true>), dim3(BT / 64, G3 / 64),
                       dim3(256), 0, stream,
                       neigh, (const _Float16*)nullptr, w_ih0_h, b_ih0,
                       (float*)nullptr, xg, BT, G3, DI_SZ);
    hipLaunchKernelGGL(gru_layer_kernel, dim3(B_SZ / 32), dim3(256), 0, stream,
                       xg, w_hh0_h, b_hh0, h0, y0, T_SZ * H_SZ);

    // ---- layer 1: xg1 = y0 @ w_ih1^T + b_ih1 ; GRU -> X0 (cols 128..) ----
    hipLaunchKernelGGL((wmma_gemm_kernel<true, true>), dim3(BT / 64, G3 / 64),
                       dim3(256), 0, stream,
                       (const float*)nullptr, y0, w_ih1_h, b_ih1,
                       (float*)nullptr, xg, BT, G3, H_SZ);
    hipLaunchKernelGGL(gru_layer_kernel, dim3(B_SZ / 32), dim3(256), 0, stream,
                       xg, w_hh1_h, b_hh1, h0 + (size_t)B_SZ * H_SZ,
                       X0 + DU_SZ, DIN_MLP);

    // ---- MLP: (X0 @ W0^T + b0) -> LN -> GELU -> ... -> d_out ----
    hipLaunchKernelGGL((wmma_gemm_kernel<true, false>), dim3(B_SZ / 64, 1024 / 64),
                       dim3(256), 0, stream,
                       (const float*)nullptr, X0, W0_h, b0,
                       Zbuf, (_Float16*)nullptr, B_SZ, 1024, DIN_MLP);
    hipLaunchKernelGGL(ln_gelu_kernel, dim3(B_SZ), dim3(256), 0, stream,
                       Zbuf, g0, be0, X1, 1024);

    hipLaunchKernelGGL((wmma_gemm_kernel<true, false>), dim3(B_SZ / 64, 512 / 64),
                       dim3(256), 0, stream,
                       (const float*)nullptr, X1, W1_h, b1,
                       Zbuf, (_Float16*)nullptr, B_SZ, 512, 1024);
    hipLaunchKernelGGL(ln_gelu_kernel, dim3(B_SZ), dim3(256), 0, stream,
                       Zbuf, g1, be1, X2, 512);

    hipLaunchKernelGGL((wmma_gemm_kernel<true, false>), dim3(B_SZ / 64, 128 / 64),
                       dim3(256), 0, stream,
                       (const float*)nullptr, X2, W2_h, b2,
                       (float*)d_out, (_Float16*)nullptr, B_SZ, 128, 512);
}
